// WindowCrossAttn_46420006535713
// MI455X (gfx1250) — matrix operand order repaired and checked by reading, not compiled, not run on hardware
//
#include <hip/hip_runtime.h>

// Fused window cross-attention for MI455X (gfx1250, wave32).
// Prep kernel: fp32 weights -> bf16, pre-swizzled into per-lane WMMA-A fragment order.
// Main kernel: one workgroup (8 waves) per 8x8 window; all GEMMs via v_wmma_f32_16x16x32_bf16;
// all fragment traffic as global_load_b128 / ds_load_b128 / ds_store_b128.

typedef __attribute__((ext_vector_type(16))) __bf16 v16bf;
typedef __attribute__((ext_vector_type(8)))  float  v8f;

union BF16x16 { v16bf v; unsigned short u[16]; uint4 q[2]; };

__device__ __forceinline__ unsigned short f2bf(float f) {
    unsigned int x = __float_as_uint(f);
    x += 0x7fffu + ((x >> 16) & 1u);                 // round-to-nearest-even
    return (unsigned short)(x >> 16);
}
__device__ __forceinline__ unsigned int pack2(float a, float b) {
    return (unsigned int)f2bf(a) | ((unsigned int)f2bf(b) << 16);
}

#define BB   8
#define CC   256
#define HH   128
#define WW   128
#define WS   8
#define NPIX 64
#define NHEAD 8
#define HD   32
#define XP   264   // row stride (ushorts) for [pixel][channel] buffers: 16B aligned, bank-rotated
#define VP   72    // row stride (ushorts) for [row][64] buffers (v, S)

#define WMMA_BF16(A, B, C) __builtin_amdgcn_wmma_f32_16x16x32_bf16(false, (A), false, (B), (short)0, (C), false, false)

// ---------------- prep: convert + swizzle weights into WMMA-A fragment order ----------------
// dst layout per matrix: frag[mtile][kblock][lane][e], e in 0..15:
//   row = mtile*16 + lane%16
//   col = kblock*32 + (e>>3)*16 + (lane>>4)*8 + (e&7)
__global__ __launch_bounds__(256)
void prep_weights_kernel(const float* __restrict__ wq, const float* __restrict__ wkv,
                         const float* __restrict__ wp, unsigned short* __restrict__ dst) {
    int i = blockIdx.x * 256 + threadIdx.x;          // 0 .. 262143
    const float* src; unsigned short* d; int li;
    if (i < 65536)       { src = wq;  d = dst;          li = i; }
    else if (i < 196608) { src = wkv; d = dst + 65536;  li = i - 65536; }
    else                 { src = wp;  d = dst + 196608; li = i - 196608; }
    int e    = li & 15;
    int lane = (li >> 4) & 31;
    int kb   = (li >> 9) & 7;
    int mt   = li >> 12;
    int row  = mt * 16 + (lane & 15);
    int col  = kb * 32 + ((e >> 3) << 4) + ((lane >> 4) << 3) + (e & 7);
    d[li] = f2bf(src[(size_t)row * CC + col]);
}

// ---------------- main fused kernel ----------------
__global__ __launch_bounds__(256)
void window_xattn_kernel(const float* __restrict__ query,
                         const float* __restrict__ kv,
                         const unsigned short* __restrict__ wfrag,   // wq | wkv | wp (bf16, swizzled)
                         const float* __restrict__ bq,
                         const float* __restrict__ bkv,
                         const float* __restrict__ bp,
                         float* __restrict__ out) {
    extern __shared__ char smem[];
    unsigned short* kT = (unsigned short*)smem;        // [p][c] 64 x 264 (k; later attn-out, col-disjoint alias)
    unsigned short* vS = kT + 64 * XP;                 // [c][p] 256 x 72
    unsigned short* XS = vS + 256 * VP;                // X staging [p][c] 64 x 264; then S: 8 x (64 x 72)

    const unsigned short* wqf  = wfrag;                // 16 mtiles
    const unsigned short* wkvf = wfrag + 65536;        // 32 mtiles
    const unsigned short* wpf  = wfrag + 196608;       // 16 mtiles

    const int tid  = threadIdx.x;
    const int lane = tid & 31;
    const int wave = tid >> 5;
    const int half = lane >> 4;
    const int l16  = lane & 15;

    const int wid = blockIdx.x;
    const int b   = wid >> 8;
    const int r0  = ((wid >> 4) & 15) * WS;
    const int c0  = (wid & 15) * WS;

    // ---- global fp32 window -> LDS bf16 [p][c]: one float4 (4 pixels of one channel) per load ----
    auto loadX = [&](const float* __restrict__ src) {
        const float* base = src + (size_t)b * CC * HH * WW;
        #pragma unroll 4
        for (int it = 0; it < 16; ++it) {
            int cid = it * 256 + tid;                 // 0..4095
            int c   = cid >> 4;                       // channel
            int j   = cid & 15;                       // chunk within window
            int r   = j >> 1, jj = j & 1;
            float4 g = *(const float4*)(base + (size_t)c * (HH * WW) +
                                        (r0 + r) * WW + c0 + jj * 4);
            int p0 = r * 8 + jj * 4;
            XS[(p0 + 0) * XP + c] = f2bf(g.x);
            XS[(p0 + 1) * XP + c] = f2bf(g.y);
            XS[(p0 + 2) * XP + c] = f2bf(g.z);
            XS[(p0 + 3) * XP + c] = f2bf(g.w);
        }
    };

    auto ld16 = [&](BF16x16& r, const unsigned short* p) {        // 16 contiguous bf16
        r.q[0] = *(const uint4*)p;
        r.q[1] = *(const uint4*)(p + 8);
    };
    auto ldA  = [&](BF16x16& r, const unsigned short* p) {        // two 8-runs at +0, +16
        r.q[0] = *(const uint4*)p;
        r.q[1] = *(const uint4*)(p + 16);
    };

    // ---- GEMM: 2 M-tiles x 4 N-tiles, K=256. A = swizzled bf16 weights (global), B = LDS [N][K] ----
    auto gemmRowPair = [&](const unsigned short* Wf, int mt0,
                           const unsigned short* Bsrc, v8f (&acc)[2][4]) {
        #pragma unroll
        for (int m = 0; m < 2; ++m)
            #pragma unroll
            for (int nt = 0; nt < 4; ++nt)
                #pragma unroll
                for (int e = 0; e < 8; ++e) acc[m][nt][e] = 0.f;
        #pragma unroll 2
        for (int k0 = 0; k0 < CC; k0 += 32) {
            BF16x16 bfr[4];
            #pragma unroll
            for (int nt = 0; nt < 4; ++nt)
                ldA(bfr[nt], Bsrc + (nt * 16 + l16) * XP + k0 + half * 8);
            #pragma unroll
            for (int m = 0; m < 2; ++m) {
                BF16x16 afr;
                const unsigned short* ap =
                    Wf + ((size_t)((mt0 + m) * 8 + (k0 >> 5)) * 32 + lane) * 16;
                afr.q[0] = *(const uint4*)ap;
                afr.q[1] = *(const uint4*)(ap + 8);
                #pragma unroll
                for (int nt = 0; nt < 4; ++nt)
                    acc[m][nt] = WMMA_BF16(afr.v, bfr[nt].v, acc[m][nt]);
            }
        }
    };

    // ================= Q projection (stays in registers: C-layout == qT A-layout) =================
    loadX(query);
    __syncthreads();
    v8f qacc[2][4];
    gemmRowPair(wqf, 2 * wave, XS, qacc);               // rows 32w..32w+31 == head w's q channels
    #pragma unroll
    for (int m = 0; m < 2; ++m)
        #pragma unroll
        for (int e = 0; e < 8; ++e) {
            float bv = bq[wave * HD + m * 16 + half * 8 + e];
            #pragma unroll
            for (int nt = 0; nt < 4; ++nt) qacc[m][nt][e] += bv;
        }
    __syncthreads();

    // ================= K,V projection =================
    loadX(kv);
    __syncthreads();
    {
        auto storeKV = [&](v8f (&acc)[2][4], int mt0) {
            #pragma unroll
            for (int m = 0; m < 2; ++m) {
                int o0 = (mt0 + m) * 16 + half * 8;
                float bv[8];
                #pragma unroll
                for (int e = 0; e < 8; ++e) bv[e] = bkv[o0 + e];
                #pragma unroll
                for (int nt = 0; nt < 4; ++nt) {
                    int p = nt * 16 + l16;
                    if ((mt0 + m) < 16) {                         // k rows -> kT[p][o], packed b128
                        uint4 pk;
                        pk.x = pack2(acc[m][nt][0] + bv[0], acc[m][nt][1] + bv[1]);
                        pk.y = pack2(acc[m][nt][2] + bv[2], acc[m][nt][3] + bv[3]);
                        pk.z = pack2(acc[m][nt][4] + bv[4], acc[m][nt][5] + bv[5]);
                        pk.w = pack2(acc[m][nt][6] + bv[6], acc[m][nt][7] + bv[7]);
                        *(uint4*)(kT + p * XP + o0) = pk;
                    } else {                                      // v rows -> vS[c][p] (transpose)
                        #pragma unroll
                        for (int e = 0; e < 8; ++e)
                            vS[(o0 - 256 + e) * VP + p] = f2bf(acc[m][nt][e] + bv[e]);
                    }
                }
            }
        };
        v8f kvacc[2][4];
        gemmRowPair(wkvf, 4 * wave, XS, kvacc);
        storeKV(kvacc, 4 * wave);
        gemmRowPair(wkvf, 4 * wave + 2, XS, kvacc);
        storeKV(kvacc, 4 * wave + 2);
    }
    __syncthreads();

    // ================= Attention: wave w owns head w =================
    const int h = wave;
    unsigned short* Sw = XS + h * (NPIX * VP);
    const float scale = 0.17677669529663687f;           // 1/sqrt(32)

    #pragma unroll
    for (int mt = 0; mt < 4; ++mt) {
        BF16x16 afr;                                    // qT tile straight from q accumulators
        #pragma unroll
        for (int e = 0; e < 8; ++e) {
            afr.u[e]     = f2bf(qacc[0][mt][e]);
            afr.u[e + 8] = f2bf(qacc[1][mt][e]);
        }
        v8f sacc[4];
        #pragma unroll
        for (int nt = 0; nt < 4; ++nt) {
            #pragma unroll
            for (int e = 0; e < 8; ++e) sacc[nt][e] = 0.f;
            BF16x16 bfr;
            ldA(bfr, kT + (nt * 16 + l16) * XP + h * HD + half * 8);
            sacc[nt] = WMMA_BF16(afr.v, bfr.v, sacc[nt]);
        }
        // row softmax: a row lives in one 16-lane half across the 4 n-tiles
        #pragma unroll
        for (int e = 0; e < 8; ++e) {
            float x0 = sacc[0][e] * scale, x1 = sacc[1][e] * scale;
            float x2 = sacc[2][e] * scale, x3 = sacc[3][e] * scale;
            float mx = fmaxf(fmaxf(x0, x1), fmaxf(x2, x3));
            mx = fmaxf(mx, __shfl_xor(mx, 1, 32));
            mx = fmaxf(mx, __shfl_xor(mx, 2, 32));
            mx = fmaxf(mx, __shfl_xor(mx, 4, 32));
            mx = fmaxf(mx, __shfl_xor(mx, 8, 32));
            x0 = __expf(x0 - mx); x1 = __expf(x1 - mx);
            x2 = __expf(x2 - mx); x3 = __expf(x3 - mx);
            float sm = x0 + x1 + x2 + x3;
            sm += __shfl_xor(sm, 1, 32);
            sm += __shfl_xor(sm, 2, 32);
            sm += __shfl_xor(sm, 4, 32);
            sm += __shfl_xor(sm, 8, 32);
            float inv = 1.0f / sm;
            int r = mt * 16 + half * 8 + e;
            Sw[r * VP +  0 + l16] = f2bf(x0 * inv);
            Sw[r * VP + 16 + l16] = f2bf(x1 * inv);
            Sw[r * VP + 32 + l16] = f2bf(x2 * inv);
            Sw[r * VP + 48 + l16] = f2bf(x3 * inv);
        }
    }

    // outT = v_h (32 x 64) @ S^T (64 x 64): A = vS rows, B = Sw rows, both K-contiguous.
    // Store packed into attnT[p][c] (aliased over kT; column range h*32.. is wave-private).
    #pragma unroll
    for (int m = 0; m < 2; ++m) {
        v8f oacc[4];
        #pragma unroll
        for (int nt = 0; nt < 4; ++nt)
            #pragma unroll
            for (int e = 0; e < 8; ++e) oacc[nt][e] = 0.f;
        #pragma unroll
        for (int ks = 0; ks < 2; ++ks) {
            int k0 = ks * 32;
            BF16x16 afr;
            ldA(afr, vS + (h * HD + m * 16 + l16) * VP + k0 + half * 8);
            #pragma unroll
            for (int nt = 0; nt < 4; ++nt) {
                BF16x16 bfr;
                ldA(bfr, Sw + (nt * 16 + l16) * VP + k0 + half * 8);
                oacc[nt] = WMMA_BF16(afr.v, bfr.v, oacc[nt]);
            }
        }
        #pragma unroll
        for (int nt = 0; nt < 4; ++nt) {
            int p = nt * 16 + l16;
            int cb = h * HD + m * 16 + half * 8;
            uint4 pk;
            pk.x = pack2(oacc[nt][0], oacc[nt][1]);
            pk.y = pack2(oacc[nt][2], oacc[nt][3]);
            pk.z = pack2(oacc[nt][4], oacc[nt][5]);
            pk.w = pack2(oacc[nt][6], oacc[nt][7]);
            *(uint4*)(kT + p * XP + cb) = pk;
        }
    }
    __syncthreads();

    // ================= Output projection -> global (window_reverse layout) =================
    {
        v8f pacc[2][4];
        gemmRowPair(wpf, 2 * wave, kT, pacc);
        #pragma unroll
        for (int m = 0; m < 2; ++m)
            #pragma unroll
            for (int e = 0; e < 8; ++e) {
                int o = (2 * wave + m) * 16 + half * 8 + e;
                float bv = bp[o];
                #pragma unroll
                for (int nt = 0; nt < 4; ++nt) {
                    int p = nt * 16 + l16;
                    out[(((size_t)b * CC + o) * HH + (r0 + (p >> 3))) * WW + (c0 + (p & 7))] =
                        pacc[m][nt][e] + bv;
                }
            }
    }
}

extern "C" void kernel_launch(void* const* d_in, const int* in_sizes, int n_in,
                              void* d_out, int out_size, void* d_ws, size_t ws_size,
                              hipStream_t stream) {
    const float* query = (const float*)d_in[0];
    const float* kvp   = (const float*)d_in[1];
    const float* wq    = (const float*)d_in[2];
    const float* bq    = (const float*)d_in[3];
    const float* wkv   = (const float*)d_in[4];
    const float* bkv   = (const float*)d_in[5];
    const float* wp    = (const float*)d_in[6];
    const float* bp    = (const float*)d_in[7];
    float* out = (float*)d_out;
    unsigned short* wfrag = (unsigned short*)d_ws;     // 262144 bf16 = 512 KB

    prep_weights_kernel<<<1024, 256, 0, stream>>>(wq, wkv, wp, wfrag);

    const size_t lds = (size_t)(64 * XP + 256 * VP + 8 * NPIX * VP) * sizeof(unsigned short); // 141 KB
    (void)hipFuncSetAttribute((const void*)window_xattn_kernel,
                              hipFuncAttributeMaxDynamicSharedMemorySize, (int)lds);
    dim3 grid(BB * (HH / WS) * (WW / WS));             // 2048 windows
    dim3 block(256);                                   // 8 waves: one head per wave
    window_xattn_kernel<<<grid, block, lds, stream>>>(query, kvp, wfrag, bq, bkv, bp, out);
}
